// TeamBall_33088428048931
// MI455X (gfx1250) — compile-verified, matrix operand order checked by reading, not run on hardware
//
#include <hip/hip_runtime.h>
#include <math.h>

typedef __attribute__((ext_vector_type(16))) _Float16 v16h;
typedef __attribute__((ext_vector_type(8)))  _Float16 v8h;
typedef __attribute__((ext_vector_type(8)))  float    v8f;

#define DEVINL __device__ __forceinline__

// ---------------- input index map (setup_inputs dict order, recursive) ----
#define IN_X          0
#define IN_T1         1
#define IN_T2         28
#define SAB1          0
#define SAB2          8
#define PMA_          16
#define M_QW 0
#define M_QB 1
#define M_KW 2
#define M_KB 3
#define M_VW 4
#define M_VB 5
#define M_OW 6
#define M_OB 7
#define TF_SEED       24
#define TF_OUTW       25
#define TF_OUTB       26
#define IN_EMB1W      55
#define IN_EMB1B      56
#define IN_EMB2W      57
#define IN_EMB2B      58
#define IN_MACRO      59
#define IN_MICRO      67
#define IN_MACRO_BACK 75
#define IN_MICRO_BACK 83
#define IN_MACRO_FCW  91
#define IN_MACRO_FCB  92
#define IN_MICRO_FCW  93
#define IN_MICRO_FCB  94
#define N_IN          95

struct Params { const float* p[N_IN]; };

// ---------------- workspace layout (bytes) --------------------------------
static const size_t OFF_E    = 0;                                // f16 [2][32768][128]
static const size_t OFF_CTX  = OFF_E   + (size_t)2*32768*128*2;  // f16 [32768][128]
static const size_t OFF_MB   = OFF_CTX + (size_t)32768*128*2;    // f32 [512][64][256]
static const size_t OFF_UB   = OFF_MB  + (size_t)32768*256*4;    // f32 [512][64][256]
static const size_t OFF_DH   = OFF_UB  + (size_t)32768*256*4;    // f16 [4][64][256]
static const size_t OFF_CC   = OFF_DH  + (size_t)4*64*256*2;     // f32 [4][64][256]
static const size_t OFF_CD   = OFF_CC  + (size_t)4*64*256*4;     // f32 [4][64][256]
static const size_t OFF_PACK = OFF_CD  + (size_t)4*64*256*4;     // packed f16 weights

// ---------------- WMMA helpers (CDNA5 16x16x32 f16, wave32) ---------------
DEVINL v8f zero8() { v8f z = {0.f,0.f,0.f,0.f,0.f,0.f,0.f,0.f}; return z; }

DEVINL v8f wmma32(v16h a, v16h b, v8f c) {
  return __builtin_amdgcn_wmma_f32_16x16x32_f16(false, a, false, b, (short)0, c, false, false);
}

// A fragment from row-major f16 [16 x ld] (LDS or global): two contiguous 128-bit loads.
// Fragment halves a[0..7]  = row m, K = k0 + 8*(lane>>4) + 0..7
//                a[8..15] = row m, K = k0 + 16 + 8*(lane>>4) + 0..7
DEVINL v16h fragA(const _Float16* base, int ld, int k0) {
  const int lane = threadIdx.x & 31;
  const int m = lane & 15, hh = lane >> 4;
  const _Float16* r = base + m*ld + k0 + hh*8;
  v8h lo = *(const v8h*)(r);
  v8h hi = *(const v8h*)(r + 16);
  v16h a;
#pragma unroll
  for (int i = 0; i < 8; ++i) { a[i] = lo[i]; a[i+8] = hi[i]; }
  return a;
}

// B fragment from pre-packed weights: one 32-byte contiguous per-lane read.
// Packed layout: tile (kt,nt) = 512 halves; lane L owns halves [L*16 .. L*16+15]
// in exact fragment order.
DEVINL v16h fragBp(const _Float16* Wp, int ntiles, int kt, int nt) {
  const int lane = threadIdx.x & 31;
  const _Float16* p = Wp + (((size_t)kt*ntiles + nt)*32 + lane)*16;
  v8h lo = *(const v8h*)(p);
  v8h hi = *(const v8h*)(p + 8);
  v16h b;
#pragma unroll
  for (int i = 0; i < 8; ++i) { b[i] = lo[i]; b[i+8] = hi[i]; }
  return b;
}

// C/D tile mapping: element (m = r + 8*(lane>>4), n = ncol + (lane&15))
DEVINL void store_tile_f16(_Float16* dst, int ld, int mbase, int ncol, const v8f& h) {
  const int lane = threadIdx.x & 31;
  const int n = ncol + (lane & 15), hh = lane >> 4;
#pragma unroll
  for (int r = 0; r < 8; ++r) dst[(size_t)(mbase + r + 8*hh) * ld + n] = (_Float16)h[r];
}
DEVINL void store_tile_f32(float* dst, int ld, int mbase, int ncol, const v8f& h) {
  const int lane = threadIdx.x & 31;
  const int n = ncol + (lane & 15), hh = lane >> 4;
#pragma unroll
  for (int r = 0; r < 8; ++r) dst[(size_t)(mbase + r + 8*hh) * ld + n] = h[r];
}
DEVINL v8f load_tile_f32(const float* src, int ld, int mbase, int ncol) {
  const int lane = threadIdx.x & 31;
  const int n = ncol + (lane & 15), hh = lane >> 4;
  v8f v;
#pragma unroll
  for (int r = 0; r < 8; ++r) v[r] = src[(size_t)(mbase + r + 8*hh) * ld + n];
  return v;
}

DEVINL float sigm(float x) { return 1.f / (1.f + __expf(-x)); }

// ================= weight pack kernel =====================================
// W[K x N] f32 row-major -> fragment-ordered f16 tiles (Kpad = ceil32(K), zero-filled)
__global__ __launch_bounds__(64) void k_pack(const float* W, int K, int N, _Float16* dst) {
  const int tile = blockIdx.x;
  const int ntiles = N >> 4;
  const int kt = tile / ntiles, nt = tile % ntiles;
  const int ebase = threadIdx.x * 8;
  v8h out;
#pragma unroll
  for (int u = 0; u < 8; ++u) {
    int e = ebase + u;
    int lane = e >> 4, j = e & 15;
    int r = j >> 1;
    int k = kt*32 + ((r & 4) ? 16 : 0) + (lane >> 4)*8 + (r & 3)*2 + (j & 1);
    int n = nt*16 + (lane & 15);
    float v = (k < K) ? W[(size_t)k*N + n] : 0.f;
    out[u] = (_Float16)v;
  }
  *(v8h*)(dst + (size_t)tile*512 + ebase) = out;
}

// ================= Stage A: set transformer ===============================
struct TfPack { const _Float16* m[2][10]; }; // s1q,s1k,s1v,s1o,s2q,s2k,s2v,s2o,pmak,pmav

// dst[16x128] = A[16xKp] @ Wp + bias (mode 0) ; dst = res + relu(acc+bias) (mode 1)
DEVINL void gemm128(const _Float16* A, int Kp, const _Float16* Wp,
                    const float* bias, _Float16* dst, const _Float16* res, int mode) {
  const int lane = threadIdx.x & 31;
  const int n_ = lane & 15, hh = lane >> 4;
  for (int nt = 0; nt < 8; ++nt) {
    v8f acc = zero8();
    for (int k0 = 0; k0 < Kp; k0 += 32)
      acc = wmma32(fragA(A, Kp, k0), fragBp(Wp, 8, k0 >> 5, nt), acc);
    int n = nt*16 + n_;
    float bn = bias[n];
#pragma unroll
    for (int r = 0; r < 8; ++r) {
      int m = r + 8*hh;
      float v = acc[r] + bn;
      if (mode == 1) v = (float)res[m*128 + n] + fmaxf(v, 0.f);
      dst[m*128 + n] = (_Float16)v;
    }
  }
}

DEVINL void sab_attn(const _Float16* qb, const _Float16* kb, const _Float16* vb,
                     float* sc, _Float16* ob) {
  const int lane = threadIdx.x & 31;
  const float scale = 0.08838834764831845f;  // 1/sqrt(128)
  for (int idx = lane; idx < 4*11*11; idx += 32) {
    int hh = idx / 121, rem = idx % 121, qi = rem / 11, kj = rem % 11;
    float acc = 0.f;
    for (int d = 0; d < 32; ++d)
      acc += (float)qb[qi*128 + hh*32 + d] * (float)kb[kj*128 + hh*32 + d];
    sc[(hh*16 + qi)*11 + kj] = acc * scale;
  }
  __syncthreads();
  for (int idx = lane; idx < 44; idx += 32) {
    int hh = idx / 11, qi = idx % 11;
    float* row = sc + (hh*16 + qi)*11;
    float mx = row[0];
    for (int j = 1; j < 11; ++j) mx = fmaxf(mx, row[j]);
    float s = 0.f;
    for (int j = 0; j < 11; ++j) { float ev = __expf(row[j] - mx); row[j] = ev; s += ev; }
    float inv = 1.f / s;
    for (int j = 0; j < 11; ++j) row[j] *= inv;
  }
  __syncthreads();
  for (int idx = lane; idx < 16*128; idx += 32) {
    int m = idx >> 7, c = idx & 127;
    float v = 0.f;
    if (m < 11) {
      int hh = c >> 5;
      v = (float)qb[m*128 + c];
      const float* row = sc + (hh*16 + m)*11;
      for (int j = 0; j < 11; ++j) v += row[j] * (float)vb[j*128 + c];
    }
    ob[idx] = (_Float16)v;
  }
  __syncthreads();
}

#define ARENA_BYTES 26624

__global__ __launch_bounds__(64) void k_set_tf(Params P, TfPack Pk, _Float16* e_ws) {
  __shared__ __align__(16) unsigned char smem[2 * ARENA_BYTES];
  const int wid = threadIdx.x >> 5, lane = threadIdx.x & 31;
  unsigned char* ar = smem + wid * ARENA_BYTES;
  _Float16* xb = (_Float16*)ar;       // [16x32]
  _Float16* qb = xb + 512;            // [16x128] each
  _Float16* kb = qb + 2048;
  _Float16* vb = kb + 2048;
  _Float16* ob = vb + 2048;
  _Float16* hb = ob + 2048;
  float*    sc = (float*)(hb + 2048); // 1280 f32

  const int item = blockIdx.x * 2 + wid;
  const int team = item >> 15;
  const int tb = item & 32767;
  const int t = tb >> 6, b = tb & 63;
  const float* const* p = P.p;
  const _Float16* const* w = Pk.m[team];
  const int base = team ? IN_T2 : IN_T1;
  const float* x = p[IN_X];

  for (int idx = lane; idx < 512; idx += 32) {
    int m = idx >> 5, k = idx & 31;
    float v = 0.f;
    if (m < 11 && k < 6) v = x[(size_t)b*512*132 + (size_t)t*132 + team*66 + m*6 + k];
    xb[idx] = (_Float16)v;
  }
  __syncthreads();
  // SAB1
  gemm128(xb, 32, w[0], p[base+SAB1+M_QB], qb, nullptr, 0);
  gemm128(xb, 32, w[1], p[base+SAB1+M_KB], kb, nullptr, 0);
  gemm128(xb, 32, w[2], p[base+SAB1+M_VB], vb, nullptr, 0);
  __syncthreads();
  sab_attn(qb, kb, vb, sc, ob);
  gemm128(ob, 128, w[3], p[base+SAB1+M_OB], hb, ob, 1);
  __syncthreads();
  // SAB2
  gemm128(hb, 128, w[4], p[base+SAB2+M_QB], qb, nullptr, 0);
  gemm128(hb, 128, w[5], p[base+SAB2+M_KB], kb, nullptr, 0);
  gemm128(hb, 128, w[6], p[base+SAB2+M_VB], vb, nullptr, 0);
  __syncthreads();
  sab_attn(qb, kb, vb, sc, ob);
  gemm128(ob, 128, w[7], p[base+SAB2+M_OB], hb, ob, 1);
  __syncthreads();
  // PMA k,v (WMMA); q/o/out are M=1 -> VALU
  gemm128(hb, 128, w[8], p[base+PMA_+M_KB], kb, nullptr, 0);
  gemm128(hb, 128, w[9], p[base+PMA_+M_VB], vb, nullptr, 0);
  const float* seed = p[base+TF_SEED];
  const float* Wq = p[base+PMA_+M_QW];
  const float* bq = p[base+PMA_+M_QB];
  float* qs = sc + 1024;
  float* vv = sc + 1152;
  for (int c = lane; c < 128; c += 32) {
    float acc = bq[c];
    for (int d = 0; d < 128; ++d) acc += seed[d] * Wq[d*128 + c];
    qs[c] = acc;
  }
  __syncthreads();
  for (int idx = lane; idx < 44; idx += 32) {
    int hh = idx / 11, j = idx % 11;
    float acc = 0.f;
    for (int d = 0; d < 32; ++d) acc += qs[hh*32 + d] * (float)kb[j*128 + hh*32 + d];
    sc[idx] = acc * 0.08838834764831845f;
  }
  __syncthreads();
  if (lane < 4) {
    float* row = sc + lane*11;
    float mx = row[0];
    for (int j = 1; j < 11; ++j) mx = fmaxf(mx, row[j]);
    float s = 0.f;
    for (int j = 0; j < 11; ++j) { float ev = __expf(row[j] - mx); row[j] = ev; s += ev; }
    float inv = 1.f / s;
    for (int j = 0; j < 11; ++j) row[j] *= inv;
  }
  __syncthreads();
  for (int c = lane; c < 128; c += 32) {
    int hh = c >> 5;
    float acc = qs[c];
    for (int j = 0; j < 11; ++j) acc += sc[hh*11 + j] * (float)vb[j*128 + c];
    vv[c] = acc;
  }
  __syncthreads();
  const float* Wo = p[base+PMA_+M_OW]; const float* bo = p[base+PMA_+M_OB];
  for (int c = lane; c < 128; c += 32) {
    float acc = bo[c];
    for (int d = 0; d < 128; ++d) acc += vv[d] * Wo[d*128 + c];
    qs[c] = vv[c] + fmaxf(acc, 0.f);
  }
  __syncthreads();
  const float* Wout = p[base+TF_OUTW]; const float* bout = p[base+TF_OUTB];
  _Float16* eo = e_ws + ((size_t)team*32768 + tb) * 128;
  for (int c = lane; c < 128; c += 32) {
    float acc = bout[c];
    for (int d = 0; d < 128; ++d) acc += qs[d] * Wout[d*128 + c];
    eo[c] = (_Float16)acc;
  }
}

// ================= Stage B: embed fc1(relu) + fc2(relu) ===================
__global__ __launch_bounds__(64) void k_embed(Params P, const _Float16* w1p, const _Float16* w2p,
                                              const _Float16* e_ws, _Float16* ctx_ws) {
  __shared__ __align__(16) _Float16 smem[2][16*256 + 16*128];
  const int wid = threadIdx.x >> 5, lane = threadIdx.x & 31;
  _Float16* eb = smem[wid];
  _Float16* hbuf = eb + 16*256;
  const int rowbase = (blockIdx.x * 2 + wid) * 16;
  for (int idx = lane; idx < 16*256; idx += 32) {
    int m = idx >> 8, c = idx & 255;
    int team = c >> 7;
    eb[idx] = e_ws[((size_t)team*32768 + rowbase + m)*128 + (c & 127)];
  }
  __syncthreads();
  const float* b1 = P.p[IN_EMB1B];
  const float* b2 = P.p[IN_EMB2B];
  const int n_ = lane & 15, hh = lane >> 4;
  for (int nt = 0; nt < 8; ++nt) {
    v8f acc = zero8();
    for (int k0 = 0; k0 < 256; k0 += 32)
      acc = wmma32(fragA(eb, 256, k0), fragBp(w1p, 8, k0 >> 5, nt), acc);
    int n = nt*16 + n_;
    float bn = b1[n];
#pragma unroll
    for (int r = 0; r < 8; ++r)
      hbuf[(r + 8*hh)*128 + n] = (_Float16)fmaxf(acc[r] + bn, 0.f);
  }
  __syncthreads();
  for (int nt = 0; nt < 8; ++nt) {
    v8f acc = zero8();
    for (int k0 = 0; k0 < 128; k0 += 32)
      acc = wmma32(fragA(hbuf, 128, k0), fragBp(w2p, 8, k0 >> 5, nt), acc);
    int n = nt*16 + n_;
    float bn = b2[n];
#pragma unroll
    for (int r = 0; r < 8; ++r)
      ctx_ws[((size_t)rowbase + r + 8*hh)*128 + n] = (_Float16)fmaxf(acc[r] + bn, 0.f);
  }
}

// ================= LSTM shared pieces =====================================
struct LstmPack { const _Float16* l0Wih; const _Float16* l0Whh;
                  const _Float16* l1Wih; const _Float16* l1Whh; };
struct BackPack { LstmPack m[2]; };

// gates over packed weight [Kpad x 1024]: ntiles = 64; cols = q*256 + nb*16
DEVINL void accum_gates(v8f* g, const _Float16* Abase, int lda, int Ksteps,
                        const _Float16* Wp, int ktBase, int nb) {
  const int lane = threadIdx.x & 31;
  for (int s = 0; s < Ksteps; ++s) {
    int kt = ktBase + s;
    v16h a = fragA(Abase, lda, s * 32);
    if (s + 1 < Ksteps)  // prefetch next K-step's B tile (global_prefetch_b8)
      __builtin_prefetch(Wp + (((size_t)(kt+1)*64 + nb)*32 + lane)*16, 0, 0);
#pragma unroll
    for (int q = 0; q < 4; ++q)
      g[q] = wmma32(a, fragBp(Wp, 64, kt, q*16 + nb), g[q]);
  }
}

DEVINL v8f lstm_cell(const v8f& g0, const v8f& g1, const v8f& g2, const v8f& g3,
                     v8f& c, const float* bih, const float* bhh, int ncol) {
  const int lane = threadIdx.x & 31;
  const int n = ncol + (lane & 15);
  float bi = bih[n]       + bhh[n];
  float bf = bih[n + 256] + bhh[n + 256];
  float bg = bih[n + 512] + bhh[n + 512];
  float bo = bih[n + 768] + bhh[n + 768];
  v8f h;
#pragma unroll
  for (int r = 0; r < 8; ++r) {
    float ig = sigm(g0[r] + bi);
    float fg = sigm(g1[r] + bf);
    float gg = tanhf(g2[r] + bg);
    float og = sigm(g3[r] + bo);
    float cv = fg * c[r] + ig * gg;
    c[r] = cv;
    h[r] = og * tanhf(cv);
  }
  return h;
}

// ================= Stage C: backward LSTMs ================================
__global__ __launch_bounds__(256) void k_back_lstm(Params P, BackPack BP, const _Float16* ctx_ws,
                                                   float* mb_ws, float* ub_ws, float* cc_ws) {
  __shared__ __align__(16) _Float16 hlds[2][64*256];   // 64 KB
  const int model = blockIdx.x;
  const int pb = model ? IN_MICRO_BACK : IN_MACRO_BACK;
  const LstmPack L = BP.m[model];
  float* outp = model ? ub_ws : mb_ws;
  float* cbase = cc_ws + (size_t)model * 2 * 64 * 256;
  const int tid = threadIdx.x, wid = tid >> 5;

  for (int i = tid; i < 2*64*256; i += 256) ((_Float16*)hlds)[i] = (_Float16)0.f;
  for (int i = tid; i < 2*64*256; i += 256) cbase[i] = 0.f;
  __threadfence();
  __syncthreads();

  for (int step = 0; step < 512; ++step) {
    const int t = 511 - step;
    v8f hnew[8];
    // ---- layer 0 ----
    for (int i = 0; i < 8; ++i) {
      int pr = wid + 8*i, mi = pr >> 4, nb = pr & 15;
      v8f g[4] = {zero8(), zero8(), zero8(), zero8()};
      accum_gates(g, ctx_ws + ((size_t)t*64 + mi*16)*128, 128, 4, L.l0Wih, 0, nb);
      accum_gates(g, hlds[0] + mi*16*256, 256, 8, L.l0Whh, 0, nb);
      v8f c = load_tile_f32(cbase, 256, mi*16, nb*16);
      hnew[i] = lstm_cell(g[0], g[1], g[2], g[3], c, P.p[pb+2], P.p[pb+3], nb*16);
      store_tile_f32(cbase, 256, mi*16, nb*16, c);
    }
    __syncthreads();
    for (int i = 0; i < 8; ++i) {
      int pr = wid + 8*i;
      store_tile_f16(hlds[0], 256, (pr >> 4)*16, (pr & 15)*16, hnew[i]);
    }
    __syncthreads();
    // ---- layer 1 ----
    for (int i = 0; i < 8; ++i) {
      int pr = wid + 8*i, mi = pr >> 4, nb = pr & 15;
      v8f g[4] = {zero8(), zero8(), zero8(), zero8()};
      accum_gates(g, hlds[0] + mi*16*256, 256, 8, L.l1Wih, 0, nb);
      accum_gates(g, hlds[1] + mi*16*256, 256, 8, L.l1Whh, 0, nb);
      v8f c = load_tile_f32(cbase + 64*256, 256, mi*16, nb*16);
      hnew[i] = lstm_cell(g[0], g[1], g[2], g[3], c, P.p[pb+6], P.p[pb+7], nb*16);
      store_tile_f32(cbase + 64*256, 256, mi*16, nb*16, c);
    }
    __syncthreads();
    for (int i = 0; i < 8; ++i) {
      int pr = wid + 8*i, mi = pr >> 4, nc = (pr & 15)*16;
      store_tile_f16(hlds[1], 256, mi*16, nc, hnew[i]);
      store_tile_f32(outp + (size_t)t*64*256, 256, mi*16, nc, hnew[i]);
    }
    __syncthreads();
  }
}

// ================= Stage D: autoregressive decoder ========================
__global__ __launch_bounds__(256) void k_decoder(Params P, LstmPack LM, LstmPack LU,
                                                 const _Float16* ctx_ws,
                                                 const float* mb_ws, const float* ub_ws,
                                                 _Float16* dh, float* cd_ws, float* out) {
  __shared__ float mp[64*2];
  __shared__ float up[64*4];
  __shared__ float fcb[64*8];
  __shared__ __align__(16) _Float16 extM[64*32];  // [ctx-tail tile for macro: mp | zeros]
  __shared__ __align__(16) _Float16 extU[64*32];  // [mp, up | zeros]
  const int tid = threadIdx.x, wid = tid >> 5;

  for (int i = tid; i < 4*64*256; i += 256) dh[i] = (_Float16)0.f;
  for (int i = tid; i < 4*64*256; i += 256) cd_ws[i] = 0.f;
  for (int i = tid; i < 128; i += 256) mp[i] = 0.f;
  for (int i = tid; i < 256; i += 256) up[i] = 0.f;
  for (int i = tid; i < 64*32; i += 256) { extM[i] = (_Float16)0.f; extU[i] = (_Float16)0.f; }
  __threadfence();
  __syncthreads();

  _Float16* hM0 = dh;
  _Float16* hM1 = dh + 64*256;
  _Float16* hU0 = dh + 2*64*256;
  _Float16* hU1 = dh + 3*64*256;
  float* cM0 = cd_ws;
  float* cM1 = cd_ws + 64*256;
  float* cU0 = cd_ws + 2*64*256;
  float* cU1 = cd_ws + 3*64*256;

  for (int t = 0; t < 512; ++t) {
    const _Float16* ctxt = ctx_ws + (size_t)t*64*128;
    v8f hnew[8];
    // ---- macro LSTM layer 0 (K = 128 ctx + extM tile; packed Kpad=160) ----
    for (int i = 0; i < 8; ++i) {
      int pr = wid + 8*i, mi = pr >> 4, nb = pr & 15;
      v8f g[4] = {zero8(), zero8(), zero8(), zero8()};
      accum_gates(g, ctxt + mi*16*128, 128, 4, LM.l0Wih, 0, nb);
      v16h ax = fragA(extM + mi*16*32, 32, 0);
#pragma unroll
      for (int q = 0; q < 4; ++q)
        g[q] = wmma32(ax, fragBp(LM.l0Wih, 64, 4, q*16 + nb), g[q]);
      accum_gates(g, hM0 + mi*16*256, 256, 8, LM.l0Whh, 0, nb);
      v8f c = load_tile_f32(cM0, 256, mi*16, nb*16);
      hnew[i] = lstm_cell(g[0], g[1], g[2], g[3], c, P.p[IN_MACRO+2], P.p[IN_MACRO+3], nb*16);
      store_tile_f32(cM0, 256, mi*16, nb*16, c);
    }
    __syncthreads();
    for (int i = 0; i < 8; ++i) { int pr = wid + 8*i;
      store_tile_f16(hM0, 256, (pr>>4)*16, (pr&15)*16, hnew[i]); }
    __threadfence(); __syncthreads();
    // ---- macro LSTM layer 1 ----
    for (int i = 0; i < 8; ++i) {
      int pr = wid + 8*i, mi = pr >> 4, nb = pr & 15;
      v8f g[4] = {zero8(), zero8(), zero8(), zero8()};
      accum_gates(g, hM0 + mi*16*256, 256, 8, LM.l1Wih, 0, nb);
      accum_gates(g, hM1 + mi*16*256, 256, 8, LM.l1Whh, 0, nb);
      v8f c = load_tile_f32(cM1, 256, mi*16, nb*16);
      hnew[i] = lstm_cell(g[0], g[1], g[2], g[3], c, P.p[IN_MACRO+6], P.p[IN_MACRO+7], nb*16);
      store_tile_f32(cM1, 256, mi*16, nb*16, c);
    }
    __syncthreads();
    for (int i = 0; i < 8; ++i) { int pr = wid + 8*i;
      store_tile_f16(hM1, 256, (pr>>4)*16, (pr&15)*16, hnew[i]); }
    __threadfence(); __syncthreads();
    // ---- macro_fc: GLU([mh, mb] @ W[512,4]) -> mp[64,2] ----
    {
      int bb = tid >> 2, gg = tid & 3;
      const float* W = P.p[IN_MACRO_FCW];
      float acc = P.p[IN_MACRO_FCB][gg];
      const _Float16* mh = hM1 + bb*256;
      const float* mb = mb_ws + ((size_t)t*64 + bb)*256;
      for (int d = 0; d < 256; ++d) acc += (float)mh[d] * W[d*4 + gg];
      for (int d = 0; d < 256; ++d) acc += mb[d] * W[(256 + d)*4 + gg];
      fcb[bb*4 + gg] = acc;
    }
    __syncthreads();
    if (tid < 128) {
      int bb = tid >> 1, gg = tid & 1;
      float v = fcb[bb*4 + gg] * sigm(fcb[bb*4 + 2 + gg]);
      mp[bb*2 + gg] = v;
      extM[bb*32 + gg] = (_Float16)v;
      extU[bb*32 + gg] = (_Float16)v;
    }
    __syncthreads();
    // ---- micro LSTM layer 0 (K = 128 ctx + extU tile; packed Kpad=160) ----
    for (int i = 0; i < 8; ++i) {
      int pr = wid + 8*i, mi = pr >> 4, nb = pr & 15;
      v8f g[4] = {zero8(), zero8(), zero8(), zero8()};
      accum_gates(g, ctxt + mi*16*128, 128, 4, LU.l0Wih, 0, nb);
      v16h ax = fragA(extU + mi*16*32, 32, 0);
#pragma unroll
      for (int q = 0; q < 4; ++q)
        g[q] = wmma32(ax, fragBp(LU.l0Wih, 64, 4, q*16 + nb), g[q]);
      accum_gates(g, hU0 + mi*16*256, 256, 8, LU.l0Whh, 0, nb);
      v8f c = load_tile_f32(cU0, 256, mi*16, nb*16);
      hnew[i] = lstm_cell(g[0], g[1], g[2], g[3], c, P.p[IN_MICRO+2], P.p[IN_MICRO+3], nb*16);
      store_tile_f32(cU0, 256, mi*16, nb*16, c);
    }
    __syncthreads();
    for (int i = 0; i < 8; ++i) { int pr = wid + 8*i;
      store_tile_f16(hU0, 256, (pr>>4)*16, (pr&15)*16, hnew[i]); }
    __threadfence(); __syncthreads();
    // ---- micro LSTM layer 1 ----
    for (int i = 0; i < 8; ++i) {
      int pr = wid + 8*i, mi = pr >> 4, nb = pr & 15;
      v8f g[4] = {zero8(), zero8(), zero8(), zero8()};
      accum_gates(g, hU0 + mi*16*256, 256, 8, LU.l1Wih, 0, nb);
      accum_gates(g, hU1 + mi*16*256, 256, 8, LU.l1Whh, 0, nb);
      v8f c = load_tile_f32(cU1, 256, mi*16, nb*16);
      hnew[i] = lstm_cell(g[0], g[1], g[2], g[3], c, P.p[IN_MICRO+6], P.p[IN_MICRO+7], nb*16);
      store_tile_f32(cU1, 256, mi*16, nb*16, c);
    }
    __syncthreads();
    for (int i = 0; i < 8; ++i) { int pr = wid + 8*i;
      store_tile_f16(hU1, 256, (pr>>4)*16, (pr&15)*16, hnew[i]); }
    __threadfence(); __syncthreads();
    // ---- micro_fc: GLU([uh, ub] @ W[512,8]) -> up[64,4] ----
    for (int o = tid; o < 512; o += 256) {
      int bb = o >> 3, gg = o & 7;
      const float* W = P.p[IN_MICRO_FCW];
      float acc = P.p[IN_MICRO_FCB][gg];
      const _Float16* uh = hU1 + bb*256;
      const float* ub = ub_ws + ((size_t)t*64 + bb)*256;
      for (int d = 0; d < 256; ++d) acc += (float)uh[d] * W[d*8 + gg];
      for (int d = 0; d < 256; ++d) acc += ub[d] * W[(256 + d)*8 + gg];
      fcb[bb*8 + gg] = acc;
    }
    __syncthreads();
    {
      int bb = tid >> 2, gg = tid & 3;
      float v = fcb[bb*8 + gg] * sigm(fcb[bb*8 + 4 + gg]);
      up[bb*4 + gg] = v;
      extU[bb*32 + 2 + gg] = (_Float16)v;
    }
    __syncthreads();
    // ---- write outputs [B,T,6] ----
    if (tid < 64) {
      float* o = out + ((size_t)tid*512 + t)*6;
      o[0] = mp[tid*2 + 0];
      o[1] = mp[tid*2 + 1];
      o[2] = up[tid*4 + 0] * 108.f;
      o[3] = up[tid*4 + 1] * 72.f;
      o[4] = up[tid*4 + 2] * 108.f;
      o[5] = up[tid*4 + 3] * 72.f;
    }
    __syncthreads();
  }
}

// ================= launch =================================================
extern "C" void kernel_launch(void* const* d_in, const int* in_sizes, int n_in,
                              void* d_out, int out_size, void* d_ws, size_t ws_size,
                              hipStream_t stream) {
  (void)in_sizes; (void)out_size; (void)ws_size;
  Params P;
  for (int i = 0; i < N_IN; ++i) P.p[i] = (i < n_in) ? (const float*)d_in[i] : nullptr;

  char* ws = (char*)d_ws;
  _Float16* e_ws   = (_Float16*)(ws + OFF_E);
  _Float16* ctx_ws = (_Float16*)(ws + OFF_CTX);
  float*    mb_ws  = (float*)(ws + OFF_MB);
  float*    ub_ws  = (float*)(ws + OFF_UB);
  _Float16* dh_ws  = (_Float16*)(ws + OFF_DH);
  float*    cc_ws  = (float*)(ws + OFF_CC);
  float*    cd_ws  = (float*)(ws + OFF_CD);
  _Float16* pkb    = (_Float16*)(ws + OFF_PACK);

  size_t off = 0;
  auto pack = [&](int pidx, int K, int N) -> const _Float16* {
    int Kp = (K + 31) & ~31;
    _Float16* dst = pkb + off;
    int tiles = (Kp >> 5) * (N >> 4);
    k_pack<<<tiles, 64, 0, stream>>>(P.p[pidx], K, N, dst);
    off += (size_t)Kp * N;
    return dst;
  };

  // set transformer weights (both teams)
  TfPack TP;
  for (int team = 0; team < 2; ++team) {
    int base = team ? IN_T2 : IN_T1;
    TP.m[team][0] = pack(base+SAB1+M_QW,   6, 128);
    TP.m[team][1] = pack(base+SAB1+M_KW,   6, 128);
    TP.m[team][2] = pack(base+SAB1+M_VW,   6, 128);
    TP.m[team][3] = pack(base+SAB1+M_OW, 128, 128);
    TP.m[team][4] = pack(base+SAB2+M_QW, 128, 128);
    TP.m[team][5] = pack(base+SAB2+M_KW, 128, 128);
    TP.m[team][6] = pack(base+SAB2+M_VW, 128, 128);
    TP.m[team][7] = pack(base+SAB2+M_OW, 128, 128);
    TP.m[team][8] = pack(base+PMA_+M_KW, 128, 128);
    TP.m[team][9] = pack(base+PMA_+M_VW, 128, 128);
  }
  const _Float16* w1p = pack(IN_EMB1W, 256, 128);
  const _Float16* w2p = pack(IN_EMB2W, 128, 128);

  BackPack BP;
  {
    int pb = IN_MACRO_BACK;
    BP.m[0] = { pack(pb+0, 128, 1024), pack(pb+1, 256, 1024),
                pack(pb+4, 256, 1024), pack(pb+5, 256, 1024) };
    pb = IN_MICRO_BACK;
    BP.m[1] = { pack(pb+0, 128, 1024), pack(pb+1, 256, 1024),
                pack(pb+4, 256, 1024), pack(pb+5, 256, 1024) };
  }
  LstmPack LM = { pack(IN_MACRO+0, 130, 1024), pack(IN_MACRO+1, 256, 1024),
                  pack(IN_MACRO+4, 256, 1024), pack(IN_MACRO+5, 256, 1024) };
  LstmPack LU = { pack(IN_MICRO+0, 134, 1024), pack(IN_MICRO+1, 256, 1024),
                  pack(IN_MICRO+4, 256, 1024), pack(IN_MICRO+5, 256, 1024) };

  k_set_tf   <<<32768, 64, 0, stream>>>(P, TP, e_ws);
  k_embed    <<<1024,  64, 0, stream>>>(P, w1p, w2p, e_ws, ctx_ws);
  k_back_lstm<<<2,    256, 0, stream>>>(P, BP, ctx_ws, mb_ws, ub_ws, cc_ws);
  k_decoder  <<<1,    256, 0, stream>>>(P, LM, LU, ctx_ws, mb_ws, ub_ws, dh_ws, cd_ws, (float*)d_out);
}
